// IndirectLearningDPD_76699525972615
// MI455X (gfx1250) — matrix operand order repaired and checked by reading, not compiled, not run on hardware
//
#include <hip/hip_runtime.h>

// CDNA5 / gfx1250: wave32, WMMA 16x16x32 f16->f32
typedef __attribute__((ext_vector_type(16))) _Float16 v16h;
typedef __attribute__((ext_vector_type(8)))  float    v8f;

#define WAVE_SAMPLES    128                  // 8 tiles of 16 samples per wave
#define WAVES_PER_BLOCK 8
#define BLOCK_SAMPLES   (WAVE_SAMPLES * WAVES_PER_BLOCK)   // 1024

// Decode basis index u -> (d, j, s).  j==0 means P==1.
//   u <16 : main terms   (j = u>>2, s = 0)
//   u <40 : lagging      (s = -lag)
//   else  : leading      (s = +lead)
// All callers pass compile-time-constant Kg so this fully constant-folds.
__device__ __forceinline__ float basis_val(int Kg,
                                           const float* xr8, const float* xi8,
                                           const float aw[3][8]) {
  const int u = Kg >> 1;          // basis column 0..63
  const int comp = Kg & 1;        // 0 = real part, 1 = imag part
  int d, j, s;
  if (u < 16)      { j = u >> 2;                 d = u & 3;  s = 0; }
  else if (u < 40) { int v = u - 16; j = (v >> 3) + 1; d = v & 3; s = -(((v >> 2) & 1) + 1); }
  else             { int v = u - 40; j = (v >> 3) + 1; d = v & 3; s =  (((v >> 2) & 1) + 1); }
  const int p  = s - d + 5;       // index into 8-wide neighborhood [-5..+2]
  const int px = 5 - d;
  const float P  = (j == 0) ? 1.0f : aw[j - 1][p];
  const float xc = comp ? xi8[px] : xr8[px];
  return xc * P;
}

// Build one 16x32 f16 A fragment (chunk C of 4, lane-half HB in {0,8}).
// 16-bit A layout (ISA 7.12.2): element e of the packed v16h maps to
// K = ((e<8)? e : e+8) + HB, HB=0 for lanes 0-15, HB=8 for lanes 16-31.
template <int C, int HB>
__device__ __forceinline__ v16h make_afrag(const float* xr8, const float* xi8,
                                           const float aw[3][8]) {
  v16h af;
#pragma unroll
  for (int e = 0; e < 16; ++e) {
    const int Kg = 32 * C + ((e < 8) ? e : e + 8) + HB;   // constant after unroll
    af[e] = (_Float16)basis_val(Kg, xr8, xi8, aw);
  }
  return af;
}

__global__ __launch_bounds__(256)
void dpd_gmp_wmma(const float* __restrict__ xr, const float* __restrict__ xi,
                  const float* __restrict__ cr, const float* __restrict__ ci,
                  float* __restrict__ out, int N) {
  const int lane = threadIdx.x & 31;
  const int half = lane >> 4;     // which 16-lane half of the wave
  const int ncol = lane & 15;     // B/D column owned by this lane
  const int wave = (int)((blockIdx.x * blockDim.x + threadIdx.x) >> 5);

  // ---- B fragments: real expansion of complex coeffs [128 x 2], cols>=2 zero.
  // Dense 16-bit B 32x16 layout (same K order as A): lane l<16 holds column l,
  // K = (e<8 ? e : e+8); lanes 16-31 hold column l&15, K = (e<8 ? e : e+8) + 8.
  v16h bfrag[4];
#pragma unroll
  for (int c = 0; c < 4; ++c) {
    v16h b = {};
    if (ncol < 2) {
#pragma unroll
      for (int e = 0; e < 16; ++e) {
        const int Kg = 32 * c + ((e < 8) ? e : e + 8) + (half ? 8 : 0);
        const int u = Kg >> 1, comp = Kg & 1;
        const float re = cr[u], im = ci[u];
        const float v = (ncol == 0) ? (comp ? -im : re)   // yr column
                                    : (comp ?  re : im);  // yi column
        b[e] = (_Float16)v;
      }
    }
    bfrag[c] = b;
  }

  const int wbase = wave * WAVE_SAMPLES;

  for (int t = 0; t < WAVE_SAMPLES / 16; ++t) {
    const int base = wbase + t * 16;
    if (base >= N) break;                       // wave-uniform: EXEC stays full
    const int row = base + ncol;                // sample this lane builds A rows for

    // ---- neighborhood x[row-5 .. row+2], zero-filled OOB; powers of |x|^2
    float xr8[8], xi8[8], aw[3][8];
#pragma unroll
    for (int p = 0; p < 8; ++p) {
      const int idx = row - 5 + p;
      const bool ok = (idx >= 0) && (idx < N);
      const float r = ok ? xr[idx] : 0.0f;
      const float m = ok ? xi[idx] : 0.0f;
      xr8[p] = r; xi8[p] = m;
      const float a  = r * r + m * m;
      const float a2 = a * a;
      aw[0][p] = a; aw[1][p] = a2; aw[2][p] = a2 * a;
    }

    // prefetch next tile's signal (global_prefetch_b8)
    if (row + 16 < N) {
      __builtin_prefetch(xr + row + 16, 0, 3);
      __builtin_prefetch(xi + row + 16, 0, 3);
    }

    // ---- A fragments: divergence only in ALU; rejoined before WMMA
    v16h af0, af1, af2, af3;
    if (half == 0) {
      af0 = make_afrag<0, 0>(xr8, xi8, aw);
      af1 = make_afrag<1, 0>(xr8, xi8, aw);
      af2 = make_afrag<2, 0>(xr8, xi8, aw);
      af3 = make_afrag<3, 0>(xr8, xi8, aw);
    } else {
      af0 = make_afrag<0, 8>(xr8, xi8, aw);
      af1 = make_afrag<1, 8>(xr8, xi8, aw);
      af2 = make_afrag<2, 8>(xr8, xi8, aw);
      af3 = make_afrag<3, 8>(xr8, xi8, aw);
    }

    // ---- D(16x16) = A(16x128) x B(128x16), K in 4 chunks of 32 (f32 accum).
    // Two independent accumulator chains to shorten the WMMA RAW chain.
    v8f acc0 = {}, acc1 = {};
    acc0 = __builtin_amdgcn_wmma_f32_16x16x32_f16(false, af0, false, bfrag[0],
                                                  (short)0, acc0, false, false);
    acc1 = __builtin_amdgcn_wmma_f32_16x16x32_f16(false, af1, false, bfrag[1],
                                                  (short)0, acc1, false, false);
    acc0 = __builtin_amdgcn_wmma_f32_16x16x32_f16(false, af2, false, bfrag[2],
                                                  (short)0, acc0, false, false);
    acc1 = __builtin_amdgcn_wmma_f32_16x16x32_f16(false, af3, false, bfrag[3],
                                                  (short)0, acc1, false, false);
    v8f acc = acc0 + acc1;

    // ---- D layout: lane l holds col l&15, rows r + (l<16 ? 0 : 8).
    // col0 = yr, col1 = yi: lanes 0/1/16/17 store their component directly
    // into the interleaved complex64 output (no cross-lane traffic).
    if (ncol < 2) {
      const int m0 = base + (half ? 8 : 0);
      if (m0 + 8 <= N) {                         // fast path: no per-row guards
#pragma unroll
        for (int r = 0; r < 8; ++r)
          out[2 * (m0 + r) + ncol] = acc[r];
      } else {
#pragma unroll
        for (int r = 0; r < 8; ++r)
          if (m0 + r < N) out[2 * (m0 + r) + ncol] = acc[r];
      }
    }
  }
}

extern "C" void kernel_launch(void* const* d_in, const int* in_sizes, int n_in,
                              void* d_out, int out_size, void* d_ws, size_t ws_size,
                              hipStream_t stream) {
  const float* xr = (const float*)d_in[0];   // x_real  [B*S]
  const float* xi = (const float*)d_in[1];   // x_imag  [B*S]
  const float* cr = (const float*)d_in[2];   // coeffs_real [64]
  const float* ci = (const float*)d_in[3];   // coeffs_imag [64]
  float* out = (float*)d_out;                // complex64 interleaved [B*S]
  const int N = in_sizes[0];
  const int blocks = (N + BLOCK_SAMPLES - 1) / BLOCK_SAMPLES;
  dpd_gmp_wmma<<<blocks, 256, 0, stream>>>(xr, xi, cr, ci, out, N);
}
